// EquiformerV2_46420006535674
// MI455X (gfx1250) — matrix operand order repaired and checked by reading, not compiled, not run on hardware
//
// EquiformerV2-style forward for MI455X (gfx1250, wave32, WMMA).
// All dense contractions lowered to v_wmma_f32_16x16x32_f16 (f16 in, f32 acc).
// Weights pre-converted to f16 *transposed* so every fragment load is a
// contiguous b128 vector load (A: two 16B runs; B^T: two 16B runs per lane).
// Irregular segment ops (softmax over incoming edges, segment-sum) via atomics.
// LDS: per-degree weight staging (ds_load_b128), radial tiles, shared-atomic
// RMS-norm reduction, FFN intermediate transpose fenced by s_wait_dscnt.
#include <hip/hip_runtime.h>
#include <math.h>

typedef _Float16 half_t;
typedef __attribute__((ext_vector_type(16))) _Float16 v16h;
typedef __attribute__((ext_vector_type(8)))  _Float16 v8h;
typedef __attribute__((ext_vector_type(8)))  float    v8f;

#define NATOMS 5000
#define NEDGES 40000
#define CCH    32
#define NLD    7
#define KCO    49
#define KC     (KCO*CCH)   // 1568
#define FF     128
#define GS     600
#define GSP    608        // padded to multiple of 32 for WMMA K
#define NH     8
#define AA     32

__constant__ int c_lidx[KCO] = {
  0,
  1,1,1,
  2,2,2,2,2,
  3,3,3,3,3,3,3,
  4,4,4,4,4,4,4,4,4,
  5,5,5,5,5,5,5,5,5,5,5,
  6,6,6,6,6,6,6,6,6,6,6,6,6};

__device__ __forceinline__ int lane_id() { return (int)(threadIdx.x & 31u); }

// A fragment (16x32 f16, row-major source): elements 0..7 <- p[k0+8*hi ..],
// elements 8..15 <- p[k0+16+8*hi ..]  (two contiguous 16B runs)
__device__ __forceinline__ v16h ld_a_frag(const half_t* row_k0, int hi) {
  v8h lo = *(const v8h*)(row_k0 + 8 * hi);
  v8h hh = *(const v8h*)(row_k0 + 16 + 8 * hi);
  return __builtin_shufflevector(lo, hh, 0,1,2,3,4,5,6,7,8,9,10,11,12,13,14,15);
}
// B fragment (32x16 f16) from transposed weights Bt[n][k]:
// elements 0..7 <- bt_col[k0+8*hi ..], elements 8..15 <- bt_col[k0+16+8*hi ..]
__device__ __forceinline__ v16h ld_b_frag(const half_t* col_k0, int hi) {
  v8h lo = *(const v8h*)(col_k0 + 8 * hi);
  v8h hh = *(const v8h*)(col_k0 + 16 + 8 * hi);
  return __builtin_shufflevector(lo, hh, 0,1,2,3,4,5,6,7,8,9,10,11,12,13,14,15);
}

__device__ __forceinline__ float fexp(float v) { return __expf(v); }
__device__ __forceinline__ float siluf(float v) {
  return v * __builtin_amdgcn_rcpf(1.0f + __expf(-v));
}

// ---------------- small utility kernels -------------------------------------
__global__ void k_fill_f32(float* p, float v, long n) {
  long i = (long)blockIdx.x * blockDim.x + threadIdx.x;
  if (i < n) p[i] = v;
}
__global__ void k_fill_u32(unsigned* p, unsigned v, long n) {
  long i = (long)blockIdx.x * blockDim.x + threadIdx.x;
  if (i < n) p[i] = v;
}
// transpose+convert: out[m][c][r] = in[m][r][c]   (rows=K, cols=N, nmat mats)
__global__ void k_cvt_t(const float* __restrict__ in, half_t* __restrict__ out,
                        int rows, int cols, long total) {
  long i = (long)blockIdx.x * blockDim.x + threadIdx.x;
  if (i >= total) return;
  long per = (long)rows * cols;
  long m = i / per, rem = i % per;
  int c = (int)(rem / rows), r = (int)(rem % rows);
  out[i] = (half_t)in[m * per + (long)r * cols + c];
}
// rad_w1 [600,128] -> f16 transposed+padded [128][608]
__global__ void k_cvt_w1t(const float* __restrict__ in, half_t* __restrict__ out) {
  long i = (long)blockIdx.x * blockDim.x + threadIdx.x;
  if (i >= (long)128 * GSP) return;
  int c = (int)(i / GSP), k = (int)(i % GSP);
  out[i] = (half_t)((k < GS) ? in[(long)k * 128 + c] : 0.0f);
}
__global__ void k_init_emb(const int* __restrict__ z, const float* __restrict__ emb,
                           float* __restrict__ x) {
  int i = blockIdx.x * blockDim.x + threadIdx.x;
  if (i >= NATOMS * CCH) return;
  int n = i >> 5, c = i & 31;
  x[(long)n * KC + c] = emb[z[n] * CCH + c];   // l=0 block only; rest stays 0
}
__global__ void k_edge_geom(const float* __restrict__ pos, const int* __restrict__ ei,
                            float* __restrict__ dist) {
  int e = blockIdx.x * blockDim.x + threadIdx.x;
  if (e >= NEDGES) return;
  int s = ei[e], d = ei[NEDGES + e];
  float dx = pos[d * 3 + 0] - pos[s * 3 + 0];
  float dy = pos[d * 3 + 1] - pos[s * 3 + 1];
  float dz = pos[d * 3 + 2] - pos[s * 3 + 2];
  dist[e] = sqrtf(dx * dx + dy * dy + dz * dz + 1e-12f);
}
__global__ void k_smear(const float* __restrict__ dist, half_t* __restrict__ sm) {
  long i = (long)blockIdx.x * blockDim.x + threadIdx.x;
  if (i >= (long)NEDGES * GSP) return;
  int e = (int)(i / GSP), g = (int)(i % GSP);
  float v = 0.0f;
  if (g < GS) {
    const float delta = 5.0f / 599.0f;
    const float co = -0.5f / ((2.0f * delta) * (2.0f * delta));
    float t = dist[e] - delta * (float)g;
    v = fexp(co * t * t);
  }
  sm[i] = (half_t)v;
}

// ---------------- per-degree RMS norm (shared-atomic reduction) -------------
__global__ void k_rmsnorm(const float* __restrict__ x, const float* __restrict__ g,
                          half_t* __restrict__ out) {
  int n = blockIdx.x;
  __shared__ float ssq[NLD];
  __shared__ float sinv[NLD];
  int tid = threadIdx.x;
  if (tid < NLD) ssq[tid] = 0.0f;
  __syncthreads();
  const float* xr = x + (long)n * KC;
  for (int idx = tid; idx < KC; idx += blockDim.x) {
    float v = xr[idx];
    atomicAdd(&ssq[c_lidx[idx >> 5]], v * v);
  }
  __syncthreads();
  if (tid < NLD) {
    float cnt = (float)((2 * tid + 1) * CCH);
    sinv[tid] = rsqrtf(ssq[tid] / cnt + 1e-5f);
  }
  __syncthreads();
  half_t* orow = out + (long)n * KC;
  for (int idx = tid; idx < KC; idx += blockDim.x) {
    int k = idx >> 5, c = idx & 31, l = c_lidx[k];
    orow[idx] = (half_t)(xr[idx] * sinv[l] * g[l * CCH + c]);
  }
}

// ---------------- generic 16x16-tile WMMA GEMM: C=act(A@Bt^T+bias) ---------
// Bt is N x K (transposed), stride = Kdim.
__global__ void __launch_bounds__(32)
k_gemm16(const half_t* __restrict__ A, int lda,
         const half_t* __restrict__ Bt,
         const float* __restrict__ bias, int act,
         half_t* __restrict__ Co, int ldc, int M, int Kdim) {
  int row0 = blockIdx.x * 16, col0 = blockIdx.y * 16;
  int ln = lane_id(), hi = ln >> 4;
  int rowA = row0 + (ln & 15); if (rowA >= M) rowA = M - 1;
  int col = col0 + (ln & 15);
  const half_t* arow = A + (long)rowA * lda;
  const half_t* bcol = Bt + (long)col * Kdim;
  v8f acc = {};
  for (int k0 = 0; k0 < Kdim; k0 += 32) {
    __builtin_prefetch(arow + k0 + 128, 0, 1);  // global_prefetch_b8
    v16h a = ld_a_frag(arow + k0, hi);
    v16h b = ld_b_frag(bcol + k0, hi);
    acc = __builtin_amdgcn_wmma_f32_16x16x32_f16(false, a, false, b, (short)0, acc, false, false);
  }
#pragma unroll
  for (int r = 0; r < 8; ++r) {
    int m = row0 + r + hi * 8;
    float v = acc[r] + (bias ? bias[col] : 0.0f);
    if (act) v = siluf(v);
    if (m < M) Co[(long)m * ldc + col] = (half_t)v;
  }
}

// ---------------- attention scalar path -------------------------------------
__global__ void k_msg0(const half_t* __restrict__ xnh, const half_t* __restrict__ rh,
                       const int* __restrict__ src, const int* __restrict__ dst,
                       half_t* __restrict__ msg0) {
  long i = (long)blockIdx.x * blockDim.x + threadIdx.x;
  if (i >= (long)NEDGES * CCH) return;
  int e = (int)(i >> 5), c = (int)(i & 31);
  int s = src[e], d = dst[e];
  float v = ((float)xnh[(long)s * KC + c] + (float)xnh[(long)d * KC + c]) *
            (float)rh[(long)e * (NLD * CCH) + c];
  msg0[i] = (half_t)v;
}

__device__ __forceinline__ unsigned enc_f32(float f) {
  unsigned u = __float_as_uint(f);
  return (u & 0x80000000u) ? ~u : (u | 0x80000000u);
}
__device__ __forceinline__ float dec_f32(unsigned u) {
  return (u & 0x80000000u) ? __uint_as_float(u & 0x7FFFFFFFu) : __uint_as_float(~u);
}

__global__ void k_logits(const half_t* __restrict__ ah, const float* __restrict__ wa2,
                         const int* __restrict__ dst, float* __restrict__ logits,
                         unsigned* __restrict__ mx) {
  int i = blockIdx.x * blockDim.x + threadIdx.x;
  if (i >= NEDGES * NH) return;
  int e = i >> 3, h = i & 7;
  const half_t* ap = ah + (long)e * (NH * AA) + h * AA;
  const float* wp = wa2 + h * AA;
  float acc = 0.0f;
#pragma unroll 8
  for (int j = 0; j < AA; ++j) acc += (float)ap[j] * wp[j];
  logits[i] = acc;
  atomicMax(&mx[dst[e] * NH + h], enc_f32(acc));
}
__global__ void k_expden(float* __restrict__ logits, const unsigned* __restrict__ mx,
                         const int* __restrict__ dst, float* __restrict__ den) {
  int i = blockIdx.x * blockDim.x + threadIdx.x;
  if (i >= NEDGES * NH) return;
  int e = i >> 3, h = i & 7;
  float ex = fexp(logits[i] - dec_f32(mx[dst[e] * NH + h]));
  logits[i] = ex;
  atomicAdd(&den[dst[e] * NH + h], ex);
}
__global__ void k_alpha(const float* __restrict__ ex, const float* __restrict__ den,
                        const int* __restrict__ dst, float* __restrict__ alpha) {
  int i = blockIdx.x * blockDim.x + threadIdx.x;
  if (i >= NEDGES * NH) return;
  int e = i >> 3, h = i & 7;
  alpha[i] = ex[i] / (den[dst[e] * NH + h] + 1e-9f);
}

// ---------------- message values: WMMA per (edge-tile,k,half) + atomic agg --
// wmsgt: transposed per-degree weights [l][d][c]
__global__ void __launch_bounds__(256)
k_msg_val(const half_t* __restrict__ xnh, const half_t* __restrict__ rh,
          const half_t* __restrict__ wmsgt, const float* __restrict__ alpha,
          const int* __restrict__ src, const int* __restrict__ dst,
          float* __restrict__ agg) {
  __shared__ alignas(32) half_t s_w[NLD * CCH * CCH];   // 14 KB transposed weights
  __shared__ alignas(32) half_t s_r[16 * NLD * CCH];    // 7 KB radial tile
  __shared__ float  s_alpha[16 * NH];
  __shared__ int    s_src[16], s_dst[16];
  int e0 = blockIdx.x * 16;
  int tid = threadIdx.x;
  for (int i = tid; i < (NLD * CCH * CCH) / 8; i += blockDim.x)
    ((v8h*)s_w)[i] = ((const v8h*)wmsgt)[i];
  for (int i = tid; i < (16 * NLD * CCH) / 8; i += blockDim.x)
    ((v8h*)s_r)[i] = ((const v8h*)(rh + (long)e0 * (NLD * CCH)))[i];
  for (int i = tid; i < 16 * NH; i += blockDim.x) s_alpha[i] = alpha[(long)e0 * NH + i];
  if (tid < 16) { s_src[tid] = src[e0 + tid]; s_dst[tid] = dst[e0 + tid]; }
  __syncthreads();

  int wid = tid >> 5, ln = tid & 31, hi = ln >> 4, mrow = ln & 15;
  int sA = s_src[mrow], dA = s_dst[mrow];
  int nwaves = (int)(blockDim.x >> 5);
  for (int idx = wid; idx < KCO * 2; idx += nwaves) {
    int k = idx >> 1, hf = idx & 1, l = c_lidx[k];
    // A = ((x1[src]+x1[dst]) * r) tile, built from b128 loads + packed f16 math
    const half_t* ps = xnh + (long)sA * KC + k * CCH;
    const half_t* pd = xnh + (long)dA * KC + k * CCH;
    const half_t* pr = s_r + mrow * (NLD * CCH) + l * CCH;
    v8h xs0 = *(const v8h*)(ps + 8 * hi),      xs1 = *(const v8h*)(ps + 16 + 8 * hi);
    v8h xd0 = *(const v8h*)(pd + 8 * hi),      xd1 = *(const v8h*)(pd + 16 + 8 * hi);
    v8h rr0 = *(const v8h*)(pr + 8 * hi),      rr1 = *(const v8h*)(pr + 16 + 8 * hi);
    v8h a0 = (xs0 + xd0) * rr0;
    v8h a1 = (xs1 + xd1) * rr1;
    v16h a = __builtin_shufflevector(a0, a1, 0,1,2,3,4,5,6,7,8,9,10,11,12,13,14,15);
    int col = hf * 16 + (ln & 15);
    v16h b = ld_b_frag(s_w + l * (CCH * CCH) + col * CCH, hi);  // ds_load_b128
    v8f acc = {};
    acc = __builtin_amdgcn_wmma_f32_16x16x32_f16(false, a, false, b, (short)0, acc, false, false);
    int head = col >> 2;  // C/H == 4 channels per head
#pragma unroll
    for (int r = 0; r < 8; ++r) {
      int m = r + hi * 8;
      float v = acc[r] * s_alpha[m * NH + head];
      atomicAdd(&agg[(long)s_dst[m] * KC + k * CCH + col], v);
    }
  }
}

// ---------------- x += agg @ w_out (per-degree, transposed weights) ---------
__global__ void __launch_bounds__(256)
k_w_out_add(const float* __restrict__ agg, const half_t* __restrict__ woutt,
            float* __restrict__ x) {
  __shared__ alignas(32) half_t s_w[NLD * CCH * CCH];
  int n0 = blockIdx.x * 16;
  int tid = threadIdx.x;
  for (int i = tid; i < (NLD * CCH * CCH) / 8; i += blockDim.x)
    ((v8h*)s_w)[i] = ((const v8h*)woutt)[i];
  __syncthreads();
  int wid = tid >> 5, ln = tid & 31, hi = ln >> 4;
  int rowA = n0 + (ln & 15); if (rowA >= NATOMS) rowA = NATOMS - 1;
  int nwaves = (int)(blockDim.x >> 5);
  for (int idx = wid; idx < KCO * 2; idx += nwaves) {
    int k = idx >> 1, hf = idx & 1, l = c_lidx[k];
    const float* pa = agg + (long)rowA * KC + k * CCH;
    v8f f0 = *(const v8f*)(pa + 8 * hi);
    v8f f1 = *(const v8f*)(pa + 16 + 8 * hi);
    v16h a;
#pragma unroll
    for (int i = 0; i < 8; ++i) { a[i] = (half_t)f0[i]; a[i + 8] = (half_t)f1[i]; }
    int col = hf * 16 + (ln & 15);
    v16h b = ld_b_frag(s_w + l * (CCH * CCH) + col * CCH, hi);
    v8f acc = {};
    acc = __builtin_amdgcn_wmma_f32_16x16x32_f16(false, a, false, b, (short)0, acc, false, false);
#pragma unroll
    for (int r = 0; r < 8; ++r) {
      int m = n0 + r + hi * 8;
      if (m < NATOMS) x[(long)m * KC + k * CCH + col] += acc[r];
    }
  }
}

// ---------------- gated FFN: x += (silu-gated (x2@wf1)) @ wf2 ---------------
// wf1t: [l][f(128)][c(32)], wf2t: [l][d(32)][f(128)]  (both transposed)
__global__ void __launch_bounds__(256)
k_ffn(const half_t* __restrict__ xnh, const half_t* __restrict__ gateh,
      const half_t* __restrict__ wf1t, const half_t* __restrict__ wf2t,
      float* __restrict__ x) {
  __shared__ alignas(32) half_t s_h[8][16 * FF];   // per-wave 16x128 intermediate
  int n0 = blockIdx.x * 16;
  int tid = threadIdx.x;
  int wid = tid >> 5, ln = tid & 31, hi = ln >> 4, nn = ln & 15;
  half_t* myh = &s_h[wid][0];
  int rowA = n0 + nn; if (rowA >= NATOMS) rowA = NATOMS - 1;
  int nwaves = (int)(blockDim.x >> 5);
  for (int k = wid; k < KCO; k += nwaves) {
    int l = c_lidx[k];
    v16h ax = ld_a_frag(xnh + (long)rowA * KC + k * CCH, hi);
    // h = x2_tile @ wf1[l], apply scalar gate, park in LDS (row-major)
#pragma unroll
    for (int ct = 0; ct < 8; ++ct) {
      int col = ct * 16 + nn;
      v16h b = ld_b_frag(wf1t + (long)l * (CCH * FF) + col * CCH, hi);
      v8f h = {};
      h = __builtin_amdgcn_wmma_f32_16x16x32_f16(false, ax, false, b, (short)0, h, false, false);
#pragma unroll
      for (int r = 0; r < 8; ++r) {
        int m = r + hi * 8;
        int rw = n0 + m; if (rw >= NATOMS) rw = NATOMS - 1;
        float g = (float)gateh[(long)rw * FF + col];
        myh[m * FF + col] = (half_t)(h[r] * g);
      }
    }
    // in-wave LDS transpose fence (CDNA5 split counters)
    asm volatile("s_wait_dscnt 0x0" ::: "memory");
    // out = h @ wf2[l]  (K=128 -> 4 chunks x 2 column halves)
    v8f z = {};
    v8f acc2[2]; acc2[0] = z; acc2[1] = z;
#pragma unroll
    for (int kc = 0; kc < 4; ++kc) {
      v16h a2 = ld_a_frag(myh + nn * FF + kc * 32, hi);   // ds_load_b128
#pragma unroll
      for (int hf = 0; hf < 2; ++hf) {
        int col = hf * 16 + nn;
        v16h b2 = ld_b_frag(wf2t + (long)l * (FF * CCH) + col * FF + kc * 32, hi);
        acc2[hf] = __builtin_amdgcn_wmma_f32_16x16x32_f16(false, a2, false, b2, (short)0,
                                                          acc2[hf], false, false);
      }
    }
#pragma unroll
    for (int hf = 0; hf < 2; ++hf) {
      int col = hf * 16 + nn;
#pragma unroll
      for (int r = 0; r < 8; ++r) {
        int m = n0 + r + hi * 8;
        if (m < NATOMS) x[(long)m * KC + k * CCH + col] += acc2[hf][r];
      }
    }
  }
}

// ---------------- host orchestration ----------------------------------------
extern "C" void kernel_launch(void* const* d_in, const int* in_sizes, int n_in,
                              void* d_out, int out_size, void* d_ws, size_t ws_size,
                              hipStream_t stream) {
  (void)in_sizes; (void)n_in; (void)out_size; (void)ws_size;
  const int*   atomic_numbers = (const int*)d_in[0];
  const float* pos     = (const float*)d_in[1];
  const int*   ei      = (const int*)d_in[2];
  const float* emb     = (const float*)d_in[3];
  const float* norm1_g = (const float*)d_in[4];
  const float* norm2_g = (const float*)d_in[5];
  const float* rad_w1  = (const float*)d_in[6];
  const float* rad_b1  = (const float*)d_in[7];
  const float* rad_w2  = (const float*)d_in[8];
  const float* rad_b2  = (const float*)d_in[9];
  const float* rad_w3  = (const float*)d_in[10];
  const float* rad_b3  = (const float*)d_in[11];
  const float* w_msg   = (const float*)d_in[12];
  const float* w_out   = (const float*)d_in[13];
  const float* wa1     = (const float*)d_in[14];
  const float* wa2     = (const float*)d_in[15];
  const float* wf1     = (const float*)d_in[16];
  const float* wg      = (const float*)d_in[17];
  const float* wf2     = (const float*)d_in[18];
  float* x = (float*)d_out;
  const int* srcp = ei;
  const int* dstp = ei + NEDGES;

  char* wsb = (char*)d_ws;
  size_t off = 0;
  auto alloc = [&](size_t bytes) -> void* {
    size_t o = (off + 255) & ~(size_t)255;
    off = o + bytes;
    return (void*)(wsb + o);
  };
  half_t* xnh   = (half_t*)alloc((size_t)NATOMS * KC * 2);
  float*  dist  = (float*) alloc((size_t)NEDGES * 4);
  half_t* smear = (half_t*)alloc((size_t)NEDGES * GSP * 2);
  half_t* w1t   = (half_t*)alloc((size_t)128 * GSP * 2);   // [128][608]
  half_t* w2t   = (half_t*)alloc((size_t)128 * 128 * 2);   // [128][128]
  half_t* w3t   = (half_t*)alloc((size_t)224 * 128 * 2);   // [224][128]
  half_t* wmsgt = (half_t*)alloc((size_t)NLD * CCH * CCH * 2);  // [l][d][c]
  half_t* woutt = (half_t*)alloc((size_t)NLD * CCH * CCH * 2);
  half_t* wa1t  = (half_t*)alloc((size_t)256 * CCH * 2);   // [256][32]
  half_t* wf1t  = (half_t*)alloc((size_t)NLD * FF * CCH * 2);   // [l][128][32]
  half_t* wf2t  = (half_t*)alloc((size_t)NLD * CCH * FF * 2);   // [l][32][128]
  half_t* wgt   = (half_t*)alloc((size_t)FF * CCH * 2);    // [128][32]
  half_t* r1h   = (half_t*)alloc((size_t)NEDGES * 128 * 2);
  half_t* r2h   = (half_t*)alloc((size_t)NEDGES * 128 * 2);
  half_t* rh    = (half_t*)alloc((size_t)NEDGES * (NLD * CCH) * 2);
  half_t* msg0h = (half_t*)alloc((size_t)NEDGES * CCH * 2);
  half_t* ahh   = (half_t*)alloc((size_t)NEDGES * 256 * 2);
  float*  logb  = (float*) alloc((size_t)NEDGES * NH * 4);
  float*  alph  = (float*) alloc((size_t)NEDGES * NH * 4);
  unsigned* mx  = (unsigned*)alloc((size_t)NATOMS * NH * 4);
  float*  den   = (float*) alloc((size_t)NATOMS * NH * 4);
  float*  agg   = (float*) alloc((size_t)NATOMS * KC * 4);
  half_t* gateh = (half_t*)alloc((size_t)NATOMS * FF * 2);

  const int BT = 256;
  auto nb = [](long n, int b) { return (unsigned)((n + b - 1) / b); };

  // ---- init x = 0, then l=0 <- emb[atomic_numbers]
  k_fill_f32<<<nb((long)NATOMS * KC, BT), BT, 0, stream>>>(x, 0.0f, (long)NATOMS * KC);
  k_init_emb<<<nb((long)NATOMS * CCH, BT), BT, 0, stream>>>(atomic_numbers, emb, x);
  k_edge_geom<<<nb(NEDGES, BT), BT, 0, stream>>>(pos, ei, dist);
  k_smear<<<nb((long)NEDGES * GSP, BT), BT, 0, stream>>>(dist, smear);

  for (int layer = 0; layer < 2; ++layer) {
    // weight conversions: f32 -> f16 transposed (w1 also K-padded 600->608)
    k_cvt_w1t<<<nb((long)128 * GSP, BT), BT, 0, stream>>>(rad_w1 + (long)layer * GS * 128, w1t);
    k_cvt_t<<<nb(128 * 128, BT), BT, 0, stream>>>(rad_w2 + (long)layer * 128 * 128, w2t, 128, 128, 128 * 128);
    k_cvt_t<<<nb(128 * 224, BT), BT, 0, stream>>>(rad_w3 + (long)layer * 128 * 224, w3t, 128, 224, 128 * 224);
    k_cvt_t<<<nb(NLD * CCH * CCH, BT), BT, 0, stream>>>(w_msg + (long)layer * NLD * CCH * CCH, wmsgt, CCH, CCH, NLD * CCH * CCH);
    k_cvt_t<<<nb(NLD * CCH * CCH, BT), BT, 0, stream>>>(w_out + (long)layer * NLD * CCH * CCH, woutt, CCH, CCH, NLD * CCH * CCH);
    k_cvt_t<<<nb(CCH * 256, BT), BT, 0, stream>>>(wa1 + (long)layer * CCH * 256, wa1t, CCH, 256, CCH * 256);
    k_cvt_t<<<nb(NLD * CCH * FF, BT), BT, 0, stream>>>(wf1 + (long)layer * NLD * CCH * FF, wf1t, CCH, FF, NLD * CCH * FF);
    k_cvt_t<<<nb(NLD * FF * CCH, BT), BT, 0, stream>>>(wf2 + (long)layer * NLD * FF * CCH, wf2t, FF, CCH, NLD * FF * CCH);
    k_cvt_t<<<nb(CCH * FF, BT), BT, 0, stream>>>(wg + (long)layer * CCH * FF, wgt, CCH, FF, CCH * FF);

    // ---- attention block ----
    k_rmsnorm<<<NATOMS, 256, 0, stream>>>(x, norm1_g + (long)layer * NLD * CCH, xnh);
    // radial MLP (WMMA GEMMs, transposed weights)
    k_gemm16<<<dim3(NEDGES / 16, 128 / 16), 32, 0, stream>>>(
        smear, GSP, w1t, rad_b1 + layer * 128, 1, r1h, 128, NEDGES, GSP);
    k_gemm16<<<dim3(NEDGES / 16, 128 / 16), 32, 0, stream>>>(
        r1h, 128, w2t, rad_b2 + layer * 128, 1, r2h, 128, NEDGES, 128);
    k_gemm16<<<dim3(NEDGES / 16, 224 / 16), 32, 0, stream>>>(
        r2h, 128, w3t, rad_b3 + layer * 224, 0, rh, 224, NEDGES, 128);
    // attention logits
    k_msg0<<<nb((long)NEDGES * CCH, BT), BT, 0, stream>>>(xnh, rh, srcp, dstp, msg0h);
    k_gemm16<<<dim3(NEDGES / 16, 256 / 16), 32, 0, stream>>>(
        msg0h, CCH, wa1t, nullptr, 1, ahh, 256, NEDGES, CCH);
    k_fill_u32<<<nb((long)NATOMS * NH, BT), BT, 0, stream>>>(mx, 0u, (long)NATOMS * NH);
    k_fill_f32<<<nb((long)NATOMS * NH, BT), BT, 0, stream>>>(den, 0.0f, (long)NATOMS * NH);
    k_logits<<<nb((long)NEDGES * NH, BT), BT, 0, stream>>>(ahh, wa2 + layer * NH * AA, dstp, logb, mx);
    k_expden<<<nb((long)NEDGES * NH, BT), BT, 0, stream>>>(logb, mx, dstp, den);
    k_alpha<<<nb((long)NEDGES * NH, BT), BT, 0, stream>>>(logb, den, dstp, alph);
    // message values + segment sum
    k_fill_f32<<<nb((long)NATOMS * KC, BT), BT, 0, stream>>>(agg, 0.0f, (long)NATOMS * KC);
    k_msg_val<<<NEDGES / 16, 256, 0, stream>>>(xnh, rh, wmsgt, alph, srcp, dstp, agg);
    k_w_out_add<<<(NATOMS + 15) / 16, 256, 0, stream>>>(agg, woutt, x);

    // ---- gated FFN ----
    k_rmsnorm<<<NATOMS, 256, 0, stream>>>(x, norm2_g + (long)layer * NLD * CCH, xnh);
    k_gemm16<<<dim3((NATOMS + 15) / 16, FF / 16), 32, 0, stream>>>(
        xnh, KC, wgt, nullptr, 1, gateh, FF, NATOMS, CCH);
    k_ffn<<<(NATOMS + 15) / 16, 256, 0, stream>>>(xnh, gateh, wf1t, wf2t, x);
  }
}
